// GMSLayer_27144193311199
// MI455X (gfx1250) — matrix operand order repaired and checked by reading, not compiled, not run on hardware
//
#include <hip/hip_runtime.h>
#include <hip/hip_bf16.h>

// ---------------- types ----------------
typedef __attribute__((ext_vector_type(16))) __bf16 v16bf;
typedef __attribute__((ext_vector_type(8)))  __bf16 v8bf;
typedef __attribute__((ext_vector_type(8)))  float  v8f;

#define E128 128

// ---------------- small helpers ----------------
__device__ __forceinline__ float sigf(float x) { return 1.0f / (1.0f + __expf(-x)); }

// Load A fragment (16x32 bf16 tile, row-major source, ld=128) in contiguous k order (k = 16h + s), same as load_b_frag:
// lane<16: K = kk+0..15 of row (lane&15); lane>=16: K = kk+16..31.
__device__ __forceinline__ v16bf load_a_frag(const __bf16* __restrict__ A, int row, int kk, int lane) {
    const int kb = kk + ((lane >> 4) << 4);
    const __bf16* p = A + (size_t)row * E128 + kb;
    v8bf lo = *reinterpret_cast<const v8bf*>(p);
    v8bf hi = *reinterpret_cast<const v8bf*>(p + 8);
    return __builtin_shufflevector(lo, hi, 0,1,2,3,4,5,6,7,8,9,10,11,12,13,14,15);
}

// Load B fragment for Y = X @ W^T. W is [N,128] row-major, so B[k][n] = W[n][k].
// 32x16 bf16 B layout: lane<16 holds K=kk+0..15 of column (lane&15); lane>=16 holds K=kk+16..31.
__device__ __forceinline__ v16bf load_b_frag(const __bf16* __restrict__ W, int col, int kk, int lane) {
    const int kb = kk + ((lane >> 4) << 4);
    return *reinterpret_cast<const v16bf*>(W + (size_t)col * E128 + kb);
}

// ---------------- conversion / elementwise kernels ----------------
__global__ void k_cvt_f32_bf16(const float* __restrict__ s, __bf16* __restrict__ d, long n) {
    long i = (long)blockIdx.x * blockDim.x + threadIdx.x;
    if (i < n) d[i] = (__bf16)s[i];
}

__global__ void k_add_f32(const float* __restrict__ a, const float* __restrict__ b,
                          float* __restrict__ o, int n) {
    int i = blockIdx.x * blockDim.x + threadIdx.x;
    if (i < n) o[i] = a[i] + b[i];
}

__global__ void k_zero_f32(float* __restrict__ p, long n) {
    long i = (long)blockIdx.x * blockDim.x + threadIdx.x;
    if (i < n) p[i] = 0.0f;
}

// comb[nv,128] f32 -> out[2*nv,128] bf16 duplicated per polarity pair
__global__ void k_dupcvt(const float* __restrict__ comb, __bf16* __restrict__ out, int nv) {
    long i = (long)blockIdx.x * blockDim.x + threadIdx.x;
    long n = (long)2 * nv * E128;
    if (i >= n) return;
    int row = (int)(i >> 7), ch = (int)(i & 127);
    int sr = (row < nv) ? row : row - nv;
    out[i] = (__bf16)comb[(size_t)sr * E128 + ch];
}

// lc_msg[j] = sum_{k<3} msg_all[lits[3j+k]]  (pure gather: clause j owns edges 3j..3j+2)
__global__ void k_gather3(const __bf16* __restrict__ msg, const int* __restrict__ lits,
                          __bf16* __restrict__ out, int C) {
    long i = (long)blockIdx.x * blockDim.x + threadIdx.x;
    long n = (long)C * E128;
    if (i >= n) return;
    int j = (int)(i >> 7), ch = (int)(i & 127);
    const int* lp = lits + (size_t)j * 3;
    float s = (float)msg[(size_t)lp[0] * E128 + ch]
            + (float)msg[(size_t)lp[1] * E128 + ch]
            + (float)msg[(size_t)lp[2] * E128 + ch];
    out[i] = (__bf16)s;
}

// clause->literal scatter: edge e (clause e/3, literal lits[e]); polarity selects cpos/cneg.
__global__ void k_scatter(const float* __restrict__ cpos, const float* __restrict__ cneg,
                          const int* __restrict__ lits, float* __restrict__ comb,
                          int nv, int C) {
    long i = (long)blockIdx.x * blockDim.x + threadIdx.x;
    long n = (long)3 * C * E128;
    if (i >= n) return;
    int e = (int)(i >> 7), ch = (int)(i & 127);
    int cl = e / 3;
    int l = lits[e];
    float v;
    float* dst;
    if (l < nv) { v = cpos[(size_t)cl * E128 + ch]; dst = comb + (size_t)l * E128 + ch; }
    else        { v = cneg[(size_t)cl * E128 + ch]; dst = comb + (size_t)(l - nv) * E128 + ch; }
    __hip_atomic_fetch_add(dst, v, __ATOMIC_RELAXED, __HIP_MEMORY_SCOPE_AGENT);
}

// ---------------- generic WMMA GEMM: Y[M,N] = act(X[M,128] @ W[N,128]^T + bias) ----------------
// One wave computes 2 row tiles x 4 col tiles. A fragments for both row tiles stay resident
// (64 VGPRs); B fragments are transient per column tile (32 VGPRs) and reused across the two
// row tiles -> ~1 global_load_b128 per WMMA at ~130 live VGPRs (no spills).
// Output mode is compile-time: no per-element branches in the store path.
template<int RELU, int OUTB, int OUTF>
__device__ __forceinline__ void store_tile(v8f acc, int m0, int col, int lane, int N,
                                           __bf16* __restrict__ outb, float* __restrict__ outf) {
    const int rbase = (lane >> 4) << 3;
#pragma unroll
    for (int r = 0; r < 8; ++r) {
        float v = acc[r];
        if (RELU) v = fmaxf(v, 0.0f);
        const size_t o = (size_t)(m0 + rbase + r) * N + col;
        if (OUTF) outf[o] = v;
        if (OUTB) outb[o] = (__bf16)v;
    }
}

template<int RELU, int OUTB, int OUTF>
__global__ __launch_bounds__(256, 1) void k_gemm(const __bf16* __restrict__ A,
                       const __bf16* __restrict__ W,
                       const float* __restrict__ bias, __bf16* __restrict__ outb,
                       float* __restrict__ outf, int M, int N) {
    const int nCG = N >> 6;
    const int nRT = M >> 4;
    const int nPairs = (nRT + 1) >> 1;
    const int total = nPairs * nCG;
    const int wave = blockIdx.x * (blockDim.x >> 5) + (threadIdx.x >> 5);
    if (wave >= total) return;               // wave-uniform: EXEC stays all-ones for WMMA
    const int lane = threadIdx.x & 31;
    const int pair = wave / nCG;
    const int cg = wave - pair * nCG;
    const int colBase = cg << 6;
    const int colIn = lane & 15;

    const int rt0 = pair * 2;
    const bool two = (rt0 + 1) < nRT;        // wave-uniform tail flag
    const int m0 = rt0 << 4;
    const int m1 = m0 + 16;
    const int rowA0 = m0 + colIn;
    const int rowA1 = (two ? m1 : m0) + colIn;   // alias row tile 0 when tail (load is unused)

    v16bf afr0[4], afr1[4];
#pragma unroll
    for (int ks = 0; ks < 4; ++ks) {
        afr0[ks] = load_a_frag(A, rowA0, ks * 32, lane);
        afr1[ks] = load_a_frag(A, rowA1, ks * 32, lane);
    }

#pragma unroll
    for (int ct = 0; ct < 4; ++ct) {
        const int col = colBase + (ct << 4) + colIn;
        v16bf bfr[4];
#pragma unroll
        for (int ks = 0; ks < 4; ++ks) bfr[ks] = load_b_frag(W, col, ks * 32, lane);
        const float bv = bias[col];
        v8f acc0, acc1;
#pragma unroll
        for (int r = 0; r < 8; ++r) { acc0[r] = bv; acc1[r] = bv; }
#pragma unroll
        for (int ks = 0; ks < 4; ++ks)
            acc0 = __builtin_amdgcn_wmma_f32_16x16x32_bf16(false, afr0[ks], false, bfr[ks],
                                                           (short)0, acc0, false, false);
        store_tile<RELU, OUTB, OUTF>(acc0, m0, col, lane, N, outb, outf);
        if (two) {
#pragma unroll
            for (int ks = 0; ks < 4; ++ks)
                acc1 = __builtin_amdgcn_wmma_f32_16x16x32_bf16(false, afr1[ks], false, bfr[ks],
                                                               (short)0, acc1, false, false);
            store_tile<RELU, OUTB, OUTF>(acc1, m1, col, lane, N, outb, outf);
        }
    }
}

// ---------------- fused LSTM cell ----------------
// g = x@Wih^T + h@Whh^T + (bih+bhh); c2 = sig(f)*c + sig(i)*tanh(g); h2 = sig(o)*tanh(c2)
// Block: 16 rows, 8 waves each owning 64 of the 512 gate columns; gates staged in LDS.
template<int HB>
__global__ __launch_bounds__(256, 1) void k_lstm(const __bf16* __restrict__ x,
                       const __bf16* __restrict__ h,
                       const __bf16* __restrict__ Wih, const __bf16* __restrict__ Whh,
                       const float* __restrict__ bias, const float* __restrict__ cin,
                       float* __restrict__ hout, float* __restrict__ cout,
                       __bf16* __restrict__ houtb, int M) {
    __shared__ float g[16 * 512];
    const int m0 = blockIdx.x << 4;
    const int wv = threadIdx.x >> 5;   // 0..7 -> column group
    const int lane = threadIdx.x & 31;
    const int row = m0 + (lane & 15);

    v16bf xa[4], ha[4];
#pragma unroll
    for (int ks = 0; ks < 4; ++ks) {
        xa[ks] = load_a_frag(x, row, ks * 32, lane);
        ha[ks] = load_a_frag(h, row, ks * 32, lane);
    }

#pragma unroll
    for (int ct = 0; ct < 4; ++ct) {
        const int n0 = (wv << 6) + (ct << 4);
        const int col = n0 + (lane & 15);
        const float binit = bias[col];
        v8f acc;
#pragma unroll
        for (int r = 0; r < 8; ++r) acc[r] = binit;
#pragma unroll
        for (int ks = 0; ks < 4; ++ks) {
            v16bf bfr = load_b_frag(Wih, col, ks * 32, lane);
            acc = __builtin_amdgcn_wmma_f32_16x16x32_bf16(false, xa[ks], false, bfr,
                                                          (short)0, acc, false, false);
        }
#pragma unroll
        for (int ks = 0; ks < 4; ++ks) {
            v16bf bfr = load_b_frag(Whh, col, ks * 32, lane);
            acc = __builtin_amdgcn_wmma_f32_16x16x32_bf16(false, ha[ks], false, bfr,
                                                          (short)0, acc, false, false);
        }
        const int rbase = (lane >> 4) << 3;
#pragma unroll
        for (int r = 0; r < 8; ++r) g[(rbase + r) * 512 + col] = acc[r];
    }
    __syncthreads();

    // Each thread owns 8 contiguous channels of one row: vector LDS/global traffic.
    {
        const int t = threadIdx.x;
        const int r = t >> 4;              // 0..15
        const int cb = (t & 15) << 3;      // 0,8,...,120
        const float* gr = &g[r * 512];
        const float4 i0 = *reinterpret_cast<const float4*>(gr + cb);
        const float4 i1 = *reinterpret_cast<const float4*>(gr + cb + 4);
        const float4 f0 = *reinterpret_cast<const float4*>(gr + 128 + cb);
        const float4 f1 = *reinterpret_cast<const float4*>(gr + 128 + cb + 4);
        const float4 g0 = *reinterpret_cast<const float4*>(gr + 256 + cb);
        const float4 g1 = *reinterpret_cast<const float4*>(gr + 256 + cb + 4);
        const float4 o0 = *reinterpret_cast<const float4*>(gr + 384 + cb);
        const float4 o1 = *reinterpret_cast<const float4*>(gr + 384 + cb + 4);
        const size_t idx = (size_t)(m0 + r) * E128 + cb;
        const float4 c0 = *reinterpret_cast<const float4*>(cin + idx);
        const float4 c1 = *reinterpret_cast<const float4*>(cin + idx + 4);

        float gi[8] = {i0.x,i0.y,i0.z,i0.w,i1.x,i1.y,i1.z,i1.w};
        float gf[8] = {f0.x,f0.y,f0.z,f0.w,f1.x,f1.y,f1.z,f1.w};
        float gg[8] = {g0.x,g0.y,g0.z,g0.w,g1.x,g1.y,g1.z,g1.w};
        float go[8] = {o0.x,o0.y,o0.z,o0.w,o1.x,o1.y,o1.z,o1.w};
        float cv[8] = {c0.x,c0.y,c0.z,c0.w,c1.x,c1.y,c1.z,c1.w};
        float c2[8], h2[8];
#pragma unroll
        for (int j = 0; j < 8; ++j) {
            c2[j] = sigf(gf[j]) * cv[j] + sigf(gi[j]) * tanhf(gg[j]);
            h2[j] = sigf(go[j]) * tanhf(c2[j]);
        }
        *reinterpret_cast<float4*>(cout + idx)     = make_float4(c2[0], c2[1], c2[2], c2[3]);
        *reinterpret_cast<float4*>(cout + idx + 4) = make_float4(c2[4], c2[5], c2[6], c2[7]);
        *reinterpret_cast<float4*>(hout + idx)     = make_float4(h2[0], h2[1], h2[2], h2[3]);
        *reinterpret_cast<float4*>(hout + idx + 4) = make_float4(h2[4], h2[5], h2[6], h2[7]);
        if (HB) {
            v8bf hb;
#pragma unroll
            for (int j = 0; j < 8; ++j) hb[j] = (__bf16)h2[j];
            *reinterpret_cast<v8bf*>(houtb + idx) = hb;
        }
    }
}

// ---------------- host orchestration ----------------
extern "C" void kernel_launch(void* const* d_in, const int* in_sizes, int n_in,
                              void* d_out, int out_size, void* d_ws, size_t ws_size,
                              hipStream_t stream) {
    const float* l_h  = (const float*)d_in[0];
    const float* l_c  = (const float*)d_in[1];
    const float* c_h  = (const float*)d_in[2];
    const float* c_c  = (const float*)d_in[3];
    const int*   edge_src = (const int*)d_in[5];   // first 3*C entries are the literal endpoints
    const float* W_lpos = (const float*)d_in[9];
    const float* b_lpos = (const float*)d_in[10];
    const float* W_lneg = (const float*)d_in[11];
    const float* b_lneg = (const float*)d_in[12];
    const float* W_cpos = (const float*)d_in[13];
    const float* b_cpos = (const float*)d_in[14];
    const float* W_cneg = (const float*)d_in[15];
    const float* b_cneg = (const float*)d_in[16];
    const float* l_Wih = (const float*)d_in[17];
    const float* l_Whh = (const float*)d_in[18];
    const float* l_bih = (const float*)d_in[19];
    const float* l_bhh = (const float*)d_in[20];
    const float* c_Wih = (const float*)d_in[21];
    const float* c_Whh = (const float*)d_in[22];
    const float* c_bih = (const float*)d_in[23];
    const float* c_bhh = (const float*)d_in[24];

    const int nl = in_sizes[0] / E128;   // 2*n_var
    const int nv = nl / 2;
    const int C  = in_sizes[2] / E128;   // n_clause
    float* out = (float*)d_out;
    float* out_lh2 = out;
    float* out_lc2 = out + (size_t)nl * E128;
    float* out_ch2 = out + (size_t)2 * nl * E128;
    float* out_cc2 = out + (size_t)2 * nl * E128 + (size_t)C * E128;

    // ---- workspace carve (256B aligned) ----
    uintptr_t cur = (uintptr_t)d_ws;
    auto alloc = [&](size_t bytes) -> void* {
        void* p = (void*)cur;
        cur += (bytes + 255) & ~(size_t)255;
        return p;
    };
    __bf16* lhbf  = (__bf16*)alloc((size_t)nl * E128 * 2);
    __bf16* chbf  = (__bf16*)alloc((size_t)C * E128 * 2);
    __bf16* wlpos = (__bf16*)alloc((size_t)3 * E128 * E128 * 2);
    __bf16* wlneg = (__bf16*)alloc((size_t)3 * E128 * E128 * 2);
    __bf16* wcpos = (__bf16*)alloc((size_t)3 * E128 * E128 * 2);
    __bf16* wcneg = (__bf16*)alloc((size_t)3 * E128 * E128 * 2);
    __bf16* lWih  = (__bf16*)alloc((size_t)512 * E128 * 2);
    __bf16* lWhh  = (__bf16*)alloc((size_t)512 * E128 * 2);
    __bf16* cWih  = (__bf16*)alloc((size_t)512 * E128 * 2);
    __bf16* cWhh  = (__bf16*)alloc((size_t)512 * E128 * 2);
    float*  lbias = (float*)alloc(512 * 4);
    float*  cbias = (float*)alloc(512 * 4);
    __bf16* tA    = (__bf16*)alloc((size_t)C * E128 * 2);   // max-M ping
    __bf16* tB    = (__bf16*)alloc((size_t)C * E128 * 2);   // max-M pong
    __bf16* msgall= (__bf16*)alloc((size_t)nl * E128 * 2);  // literal messages; later aliased by comb
    __bf16* lcbf  = (__bf16*)alloc((size_t)C * E128 * 2);   // clause input msgs; later aliased by clbf
    __bf16* ch2bf = (__bf16*)alloc((size_t)C * E128 * 2);
    float*  cposm = (float*)alloc((size_t)C * E128 * 4);
    float*  cnegm = (float*)alloc((size_t)C * E128 * 4);
    float*  comb  = (float*)msgall;   // nv*128 f32 == nl*128 bf16 bytes; msgall dead after gather
    __bf16* clbf  = lcbf;             // nl*128 <= C*128; lcbf dead after clause LSTM

    auto cvt = [&](const float* s, __bf16* d, long n) {
        k_cvt_f32_bf16<<<(int)((n + 255) / 256), 256, 0, stream>>>(s, d, n);
    };
    auto gblocks = [&](int M, int N) {
        const int waves = (((M >> 4) + 1) >> 1) * (N >> 6);
        return (waves + 7) / 8;
    };
    auto gemmRB = [&](const __bf16* A, const __bf16* W, const float* bias, __bf16* ob, int M) {
        k_gemm<1,1,0><<<gblocks(M, E128), 256, 0, stream>>>(A, W, bias, ob, nullptr, M, E128);
    };
    auto gemmB = [&](const __bf16* A, const __bf16* W, const float* bias, __bf16* ob, int M) {
        k_gemm<0,1,0><<<gblocks(M, E128), 256, 0, stream>>>(A, W, bias, ob, nullptr, M, E128);
    };
    auto gemmF = [&](const __bf16* A, const __bf16* W, const float* bias, float* of, int M) {
        k_gemm<0,0,1><<<gblocks(M, E128), 256, 0, stream>>>(A, W, bias, nullptr, of, M, E128);
    };

    // ---- precision prep ----
    cvt(W_lpos, wlpos, 3L * E128 * E128);  cvt(W_lneg, wlneg, 3L * E128 * E128);
    cvt(W_cpos, wcpos, 3L * E128 * E128);  cvt(W_cneg, wcneg, 3L * E128 * E128);
    cvt(l_Wih, lWih, 512L * E128);         cvt(l_Whh, lWhh, 512L * E128);
    cvt(c_Wih, cWih, 512L * E128);         cvt(c_Whh, cWhh, 512L * E128);
    k_add_f32<<<2, 256, 0, stream>>>(l_bih, l_bhh, lbias, 512);
    k_add_f32<<<2, 256, 0, stream>>>(c_bih, c_bhh, cbias, 512);
    cvt(l_h, lhbf, (long)nl * E128);
    cvt(c_h, chbf, (long)C * E128);

    // ---- literal MLPs (per polarity, 3 layers, ReLU between) ----
    gemmRB(lhbf,                     wlpos,         b_lpos,       tA,     nv);
    gemmRB(tA,                       wlpos + 16384, b_lpos + 128, tB,     nv);
    gemmB (tB,                       wlpos + 32768, b_lpos + 256, msgall, nv);
    gemmRB(lhbf + (size_t)nv * E128, wlneg,         b_lneg,       tA,     nv);
    gemmRB(tA,                       wlneg + 16384, b_lneg + 128, tB,     nv);
    gemmB (tB,                       wlneg + 32768, b_lneg + 256, msgall + (size_t)nv * E128, nv);

    // ---- literal -> clause aggregation (pure gather of 3) ----
    {
        long n = (long)C * E128;
        k_gather3<<<(int)((n + 255) / 256), 256, 0, stream>>>(msgall, edge_src, lcbf, C);
    }

    // ---- clause LSTM (writes c_h2/c_c2 to d_out; keeps bf16 copy of c_h2) ----
    k_lstm<1><<<C / 16, 256, 0, stream>>>(lcbf, chbf, cWih, cWhh, cbias, c_c,
                                          out_ch2, out_cc2, ch2bf, C);

    // ---- clause MLPs (pos/neg), final layer in f32 for the atomic scatter ----
    gemmRB(ch2bf, wcpos,         b_cpos,       tA,    C);
    gemmRB(tA,    wcpos + 16384, b_cpos + 128, tB,    C);
    gemmF (tB,    wcpos + 32768, b_cpos + 256, cposm, C);
    gemmRB(ch2bf, wcneg,         b_cneg,       tA,    C);
    gemmRB(tA,    wcneg + 16384, b_cneg + 128, tB,    C);
    gemmF (tB,    wcneg + 32768, b_cneg + 256, cnegm, C);

    // ---- clause -> literal aggregation (atomic scatter), paired combine, duplicate ----
    {
        long n = (long)nv * E128;
        k_zero_f32<<<(int)((n + 255) / 256), 256, 0, stream>>>(comb, n);
        long ne = (long)3 * C * E128;
        k_scatter<<<(int)((ne + 255) / 256), 256, 0, stream>>>(cposm, cnegm, edge_src, comb, nv, C);
        long nd = (long)nl * E128;
        k_dupcvt<<<(int)((nd + 255) / 256), 256, 0, stream>>>(comb, clbf, nv);
    }

    // ---- literal LSTM (writes l_h2/l_c2 to d_out) ----
    k_lstm<0><<<nl / 16, 256, 0, stream>>>(clbf, lhbf, lWih, lWhh, lbias, l_c,
                                           out_lh2, out_lc2, nullptr, nl);
}
